// SequentialLlama4TextMoe_86019605004825
// MI455X (gfx1250) — matrix operand order repaired and checked by reading, not compiled, not run on hardware
//
#include <hip/hip_runtime.h>
#include <hip/hip_bf16.h>

// ---------------------------------------------------------------------------
// Llama4-style MoE (dense experts, top-1 router) for gfx1250 (MI455X).
// Compute-bound (~0.93 TFLOP vs ~0.9 GB weights): bf16 WMMA (16x16x32) with
// f32 accumulation. X is pre-converted to bf16 once; bf16 tiles (X, P) are
// staged into LDS with global_load_async_to_lds_b128 (ASYNCcnt); fp32 weight
// tiles are converted in-flight with v_perm_b32 packing.
// ---------------------------------------------------------------------------

typedef __attribute__((ext_vector_type(16))) __bf16    v16bf;
typedef __attribute__((ext_vector_type(8)))  float     v8f;
typedef __attribute__((ext_vector_type(4)))  float     v4f;
typedef __attribute__((ext_vector_type(4)))  unsigned  v4u;
typedef __attribute__((ext_vector_type(2)))  unsigned  v2u;

#define T_DIM 2048
#define H_DIM 2048
#define I_DIM 4096
#define E_NUM 8

#define BM 128      // token tile
#define BN 64       // output-feature tile
#define BK 32       // K tile == WMMA K depth for bf16
#define LDT 40      // LDS row stride in halves (BK + 8 pad -> 80B rows)

union FragB16 { v16bf v; v4u q[2]; };

// One v_perm_b32: [b.hi16, a.hi16] (truncating fp32->bf16 pack).
__device__ __forceinline__ unsigned pack_bf16(float a, float b) {
  union { float f; unsigned u; } ua, ub; ua.f = a; ub.f = b;
  return __builtin_amdgcn_perm(ub.u, ua.u, 0x07060302u);
}

__device__ __forceinline__ unsigned short to_bf16(float a) {
  union { float f; unsigned u; } c; c.f = a;
  return (unsigned short)(c.u >> 16);
}

// Async copy 16B global -> LDS (tracked by ASYNCcnt, no VGPR round trip).
__device__ __forceinline__ void async_load_b128(void* lds_ptr, const void* gptr) {
  unsigned lds = (unsigned)(uintptr_t)lds_ptr;              // low 32b = LDS offset
  unsigned long long ga = (unsigned long long)(uintptr_t)gptr;
  asm volatile("global_load_async_to_lds_b128 %0, %1, off"
               :: "v"(lds), "v"(ga) : "memory");
}

__device__ __forceinline__ void wait_async() {
  asm volatile("s_wait_asynccnt 0x0" ::: "memory");
}

// A fragment (16x32, M x K). Lane L<16 holds row M=L: K {0..7} in q0, {16..23}
// in q1; lane 16+L holds K {8..15} and {24..31}.  (ISA 7.12.2, 16-bit A)
__device__ __forceinline__ v16bf load_a_frag(const unsigned short* s, int lane) {
  const unsigned short* base = s + (lane & 15) * LDT;
  const int kb = (lane >> 4) << 3;                 // 0 or 8 halves
  FragB16 f;
  f.q[0] = *reinterpret_cast<const v4u*>(base + kb);
  f.q[1] = *reinterpret_cast<const v4u*>(base + kb + 16);
  return f.v;
}

// B fragment (32x16, K x N) built from W stored row-major [N][K] in LDS:
// lane L<16 -> column N=L, K {0..7},{16..23}; lane 16+L -> K {8..15},{24..31}.
__device__ __forceinline__ v16bf load_b_frag(const unsigned short* s, int lane) {
  const unsigned short* base = s + (lane & 15) * LDT;
  const int kb = (lane >> 4) << 3;                 // 0 or 8 halves
  FragB16 f;
  f.q[0] = *reinterpret_cast<const v4u*>(base + kb);
  f.q[1] = *reinterpret_cast<const v4u*>(base + kb + 16);
  return f.v;
}

__device__ __forceinline__ v8f wmma_bf16(v16bf a, v16bf b, v8f c) {
  return __builtin_amdgcn_wmma_f32_16x16x32_bf16(
      /*neg_a=*/false, a, /*neg_b=*/false, b,
      /*c_mod=*/(short)0, c, /*reuse_a=*/false, /*reuse_b=*/false);
}

__device__ __forceinline__ float silu_f(float x) {
  return x / (1.0f + __expf(-x));
}

// ---------------------------------------------------------------------------
// X (fp32) -> Xb (bf16), done once per launch. 8 floats / thread.
// ---------------------------------------------------------------------------
__global__ __launch_bounds__(256)
void cvt_x_kernel(const float* __restrict__ X, unsigned short* __restrict__ Xb) {
  size_t i = ((size_t)blockIdx.x * 256 + threadIdx.x) * 8;
  v4f a = *reinterpret_cast<const v4f*>(X + i);
  v4f b = *reinterpret_cast<const v4f*>(X + i + 4);
  v4u o;
  o.x = pack_bf16(a.x, a.y); o.y = pack_bf16(a.z, a.w);
  o.z = pack_bf16(b.x, b.y); o.w = pack_bf16(b.z, b.w);
  *reinterpret_cast<v4u*>(Xb + i) = o;
}

// ---------------------------------------------------------------------------
// Router: logits = x @ router_w^T, keep top-1, sigmoid; scores [E][T].
// One wave per token; 8 waves / block.
// ---------------------------------------------------------------------------
__global__ __launch_bounds__(256)
void router_kernel(const float* __restrict__ x, const float* __restrict__ rw,
                   float* __restrict__ scores) {
  const int lane = threadIdx.x & 31;
  const int wave = threadIdx.x >> 5;
  const int t = blockIdx.x * 8 + wave;

  float acc[E_NUM];
#pragma unroll
  for (int e = 0; e < E_NUM; ++e) acc[e] = 0.0f;

  const float* xrow = x + (size_t)t * H_DIM;
  for (int h = lane; h < H_DIM; h += 32) {
    float xv = xrow[h];
#pragma unroll
    for (int e = 0; e < E_NUM; ++e)
      acc[e] += xv * rw[(size_t)e * H_DIM + h];
  }
#pragma unroll
  for (int e = 0; e < E_NUM; ++e) {
#pragma unroll
    for (int off = 16; off > 0; off >>= 1)
      acc[e] += __shfl_xor(acc[e], off, 32);
  }
  if (lane == 0) {
    int best = 0; float bv = acc[0];
#pragma unroll
    for (int e = 1; e < E_NUM; ++e)
      if (acc[e] > bv) { bv = acc[e]; best = e; }
    const float sig = 1.0f / (1.0f + __expf(-bv));
#pragma unroll
    for (int e = 0; e < E_NUM; ++e)
      scores[(size_t)e * T_DIM + t] = (e == best) ? sig : 0.0f;
  }
}

// ---------------------------------------------------------------------------
// Gate+Up: P[T,I] = silu(Xb Wg^T) * (Xb Wu^T) for one expert (bf16 -> ws).
// Block: 256 threads = 8 waves (4 M-slices x 2 N-slices), tile 128x64.
// A-tile: async bf16 copy.  W-tiles: fp32 -> bf16 via v_perm.
// ---------------------------------------------------------------------------
__global__ __launch_bounds__(256)
void gateup_kernel(const unsigned short* __restrict__ Xb,
                   const float* __restrict__ Wg, const float* __restrict__ Wu,
                   unsigned short* __restrict__ P) {
  __shared__ __align__(16) unsigned short sA[BM * LDT];
  __shared__ __align__(16) unsigned short sG[BN * LDT];
  __shared__ __align__(16) unsigned short sU[BN * LDT];

  const int tid   = threadIdx.x;
  const int lane  = tid & 31;
  const int wave  = tid >> 5;
  const int waveM = wave & 3;          // 4 x 32-row slices
  const int waveN = wave >> 2;         // 2 x 32-col slices
  const int m0 = blockIdx.y * BM;
  const int n0 = blockIdx.x * BN;

  v8f ag[2][2] = {}; v8f au[2][2] = {};

  for (int kt = 0; kt < H_DIM; kt += BK) {
    // Async-stage Xb tile (128x32 bf16): 512 16B chunks, 2 per thread.
#pragma unroll
    for (int j = 0; j < 2; ++j) {
      int idx = tid + j * 256;
      int r = idx >> 2, c = idx & 3;
      async_load_b128(&sA[r * LDT + c * 8],
                      Xb + (size_t)(m0 + r) * H_DIM + kt + c * 8);
    }
    // Stage Wg/Wu tiles (64x32 fp32 -> bf16 each): 512 float4 chunks, 2/thread.
#pragma unroll
    for (int j = 0; j < 2; ++j) {
      int idx = tid + j * 256;
      int r = idx >> 3, c4 = idx & 7;
      size_t go = (size_t)(n0 + r) * H_DIM + kt + c4 * 4;
      v4f vg = *reinterpret_cast<const v4f*>(Wg + go);
      v4f vu = *reinterpret_cast<const v4f*>(Wu + go);
      v2u pg; pg.x = pack_bf16(vg.x, vg.y); pg.y = pack_bf16(vg.z, vg.w);
      v2u pu; pu.x = pack_bf16(vu.x, vu.y); pu.y = pack_bf16(vu.z, vu.w);
      *reinterpret_cast<v2u*>(&sG[r * LDT + c4 * 4]) = pg;
      *reinterpret_cast<v2u*>(&sU[r * LDT + c4 * 4]) = pu;
    }
    if (kt + BK < H_DIM)                        // global_prefetch_b8 hint
      __builtin_prefetch(Wg + (size_t)(n0 + (tid >> 3)) * H_DIM + kt + BK, 0, 1);
    wait_async();
    __syncthreads();

    v16bf af[2], bg[2], bu[2];
#pragma unroll
    for (int mf = 0; mf < 2; ++mf)
      af[mf] = load_a_frag(sA + (waveM * 32 + mf * 16) * LDT, lane);
#pragma unroll
    for (int nf = 0; nf < 2; ++nf) {
      bg[nf] = load_b_frag(sG + (waveN * 32 + nf * 16) * LDT, lane);
      bu[nf] = load_b_frag(sU + (waveN * 32 + nf * 16) * LDT, lane);
    }
#pragma unroll
    for (int mf = 0; mf < 2; ++mf)
#pragma unroll
      for (int nf = 0; nf < 2; ++nf) {
        ag[mf][nf] = wmma_bf16(af[mf], bg[nf], ag[mf][nf]);
        au[mf][nf] = wmma_bf16(af[mf], bu[nf], au[mf][nf]);
      }
    __syncthreads();
  }

  // Epilogue: P = silu(g) * u, stored bf16.
  const int rhalf = (lane >> 4) * 8;
  const int cl = lane & 15;
#pragma unroll
  for (int mf = 0; mf < 2; ++mf)
#pragma unroll
    for (int nf = 0; nf < 2; ++nf) {
#pragma unroll
      for (int r = 0; r < 8; ++r) {
        float p = silu_f(ag[mf][nf][r]) * au[mf][nf][r];
        int row = m0 + waveM * 32 + mf * 16 + rhalf + r;
        int col = n0 + waveN * 32 + nf * 16 + cl;
        P[(size_t)row * I_DIM + col] = to_bf16(p);
      }
    }
}

// ---------------------------------------------------------------------------
// Down: out (+)= score .* (P Wd^T).  INIT: overwrite (shared expert, score=1).
// P-tile: async bf16 copy.  Wd-tile: fp32 -> bf16 via v_perm.
// ---------------------------------------------------------------------------
template<bool INIT, bool USE_SCORE>
__global__ __launch_bounds__(256)
void down_kernel(const unsigned short* __restrict__ P, const float* __restrict__ Wd,
                 const float* __restrict__ S, float* __restrict__ out) {
  __shared__ __align__(16) unsigned short sA[BM * LDT];
  __shared__ __align__(16) unsigned short sB[BN * LDT];

  const int tid   = threadIdx.x;
  const int lane  = tid & 31;
  const int wave  = tid >> 5;
  const int waveM = wave & 3;
  const int waveN = wave >> 2;
  const int m0 = blockIdx.y * BM;
  const int n0 = blockIdx.x * BN;

  v8f acc[2][2] = {};

  for (int kt = 0; kt < I_DIM; kt += BK) {
    // Async-stage P tile (128x32 bf16): 512 16B chunks, 2 per thread.
#pragma unroll
    for (int j = 0; j < 2; ++j) {
      int idx = tid + j * 256;
      int r = idx >> 2, c = idx & 3;
      async_load_b128(&sA[r * LDT + c * 8],
                      P + (size_t)(m0 + r) * I_DIM + kt + c * 8);
    }
    // Stage Wd tile (64x32 fp32 -> bf16).
#pragma unroll
    for (int j = 0; j < 2; ++j) {
      int idx = tid + j * 256;
      int r = idx >> 3, c4 = idx & 7;
      v4f v = *reinterpret_cast<const v4f*>(Wd + (size_t)(n0 + r) * I_DIM + kt + c4 * 4);
      v2u pk; pk.x = pack_bf16(v.x, v.y); pk.y = pack_bf16(v.z, v.w);
      *reinterpret_cast<v2u*>(&sB[r * LDT + c4 * 4]) = pk;
    }
    if (kt + BK < I_DIM)
      __builtin_prefetch(Wd + (size_t)(n0 + (tid >> 3)) * I_DIM + kt + BK, 0, 1);
    wait_async();
    __syncthreads();

    v16bf af[2], bf[2];
#pragma unroll
    for (int mf = 0; mf < 2; ++mf)
      af[mf] = load_a_frag(sA + (waveM * 32 + mf * 16) * LDT, lane);
#pragma unroll
    for (int nf = 0; nf < 2; ++nf)
      bf[nf] = load_b_frag(sB + (waveN * 32 + nf * 16) * LDT, lane);
#pragma unroll
    for (int mf = 0; mf < 2; ++mf)
#pragma unroll
      for (int nf = 0; nf < 2; ++nf)
        acc[mf][nf] = wmma_bf16(af[mf], bf[nf], acc[mf][nf]);
    __syncthreads();
  }

  const int rhalf = (lane >> 4) * 8;
  const int cl = lane & 15;
#pragma unroll
  for (int mf = 0; mf < 2; ++mf)
#pragma unroll
    for (int r = 0; r < 8; ++r) {
      int row = m0 + waveM * 32 + mf * 16 + rhalf + r;
      float s = USE_SCORE ? S[row] : 1.0f;
#pragma unroll
      for (int nf = 0; nf < 2; ++nf) {
        int col = n0 + waveN * 32 + nf * 16 + cl;
        size_t o = (size_t)row * H_DIM + col;
        float val = acc[mf][nf][r] * s;
        if (INIT) out[o] = val;
        else      out[o] += val;
      }
    }
}

// ---------------------------------------------------------------------------
extern "C" void kernel_launch(void* const* d_in, const int* in_sizes, int n_in,
                              void* d_out, int out_size, void* d_ws, size_t ws_size,
                              hipStream_t stream) {
  (void)in_sizes; (void)n_in; (void)out_size; (void)ws_size;

  const float* x  = (const float*)d_in[0];   // [T, H]
  const float* rw = (const float*)d_in[1];   // [E, H]
  const float* eg = (const float*)d_in[2];   // [E, I, H]
  const float* eu = (const float*)d_in[3];   // [E, I, H]
  const float* ed = (const float*)d_in[4];   // [E, H, I]
  const float* sg = (const float*)d_in[5];   // [I, H]
  const float* su = (const float*)d_in[6];   // [I, H]
  const float* sd = (const float*)d_in[7];   // [H, I]

  float* out    = (float*)d_out;                         // [T, H]
  float* scores = out + (size_t)T_DIM * H_DIM;           // [E, T]
  unsigned short* P  = (unsigned short*)d_ws;            // [T, I]  bf16 (16 MB)
  unsigned short* Xb = P + (size_t)T_DIM * I_DIM;        // [T, H]  bf16 ( 8 MB)

  const dim3 blk(256);
  const dim3 gGU(I_DIM / BN, T_DIM / BM);   // 64 x 16
  const dim3 gDN(H_DIM / BN, T_DIM / BM);   // 32 x 16

  router_kernel<<<T_DIM / 8, blk, 0, stream>>>(x, rw, scores);
  cvt_x_kernel<<<(T_DIM * H_DIM) / (256 * 8), blk, 0, stream>>>(x, Xb);

  // Shared expert initializes out (score = 1).
  gateup_kernel<<<gGU, blk, 0, stream>>>(Xb, sg, su, P);
  down_kernel<true, false><<<gDN, blk, 0, stream>>>(P, sd, nullptr, out);

  // Dense experts, masked by router scores, accumulate.
  for (int e = 0; e < E_NUM; ++e) {
    const size_t wo = (size_t)e * I_DIM * H_DIM;
    gateup_kernel<<<gGU, blk, 0, stream>>>(Xb, eg + wo, eu + wo, P);
    down_kernel<false, true><<<gDN, blk, 0, stream>>>(
        P, ed + wo, scores + (size_t)e * T_DIM, out);
  }
}
